// MultiHeadAttention_31963146617272
// MI455X (gfx1250) — matrix operand order repaired and checked by reading, not compiled, not run on hardware
//
#include <hip/hip_runtime.h>
#include <math.h>

typedef _Float16 v8h  __attribute__((ext_vector_type(8)));
typedef _Float16 v16h __attribute__((ext_vector_type(16)));
typedef float    v8f  __attribute__((ext_vector_type(8)));

#define N_STATE 1024
#define N_HEAD  16
#define HD      64
#define BATCH   2
#define SEQ     2048
#define M_TOT   (BATCH*SEQ)

// ---- WMMA fragment loaders (16x16x32 f16, contiguous k order used by both A and B) ----
// A (16x32, MxK): lane(l,hi) holds row M=l; halves i -> K=hi*16+i
//                 => two contiguous 16B chunks (one 32B run).
static __device__ __forceinline__ v16h load_a_frag(const _Float16* base, int ld,
                                                   int l, int hi) {
  union { v16h v; v8h h[2]; } a;
  a.h[0] = *(const v8h*)(base + (size_t)l*ld + hi*16);
  a.h[1] = *(const v8h*)(base + (size_t)l*ld + hi*16 + 8);
  return a.v;
}
// B (32x16, KxN): lane(l,hi) holds col N=l; halves i -> K=hi*16+i
//                 => one contiguous 32B chunk from row l of the KxN-as-rows src.
static __device__ __forceinline__ v16h load_b_frag(const _Float16* base, int ld,
                                                   int l, int hi) {
  return *(const v16h*)(base + (size_t)l*ld + hi*16);
}
static __device__ __forceinline__ v8f wmma_f16(v16h a, v16h b, v8f c) {
  return __builtin_amdgcn_wmma_f32_16x16x32_f16(false, a, false, b,
                                                (short)0, c, false, false);
}

// ---------------- fp32 -> fp16 convert (with optional scale fold) ----------
__global__ void cvt_f32_to_f16(const float* __restrict__ src,
                               _Float16* __restrict__ dst, int n, float scale) {
  int i = blockIdx.x*blockDim.x + threadIdx.x;
  if (i < n) dst[i] = (_Float16)(src[i]*scale);
}

// ---------------- fused Q/K/V projection (z selects output) ----------------
// out[m][n] = sum_k xh[m][k]*W[n][k] (+bias).  Wave tile = 64x64.
// q,k: row-major f16 [4096][1024].  v: stored transposed vT[b][h][d][t].
__global__ __launch_bounds__(128)
void qkv_proj_kernel(const _Float16* __restrict__ xh,
                     const _Float16* __restrict__ Wqh,
                     const _Float16* __restrict__ Wkh,
                     const _Float16* __restrict__ Wvh,
                     const float* __restrict__ bq,
                     const float* __restrict__ bv,
                     _Float16* __restrict__ qh,
                     _Float16* __restrict__ kh,
                     _Float16* __restrict__ vT) {
  const int lane = threadIdx.x & 31, wave = threadIdx.x >> 5;
  const int l = lane & 15, hi = lane >> 4;
  const int z  = blockIdx.z;
  const int m0 = (blockIdx.x*4 + wave)*64;
  const int n0 = blockIdx.y*64;
  const _Float16* W = (z==0) ? Wqh : (z==1) ? Wkh : Wvh;
  const float QSCALE = 0.35355339059327379f;   // 64^(-1/4); Wq/Wk pre-scaled at cvt

  v8f acc[4][4];
  for (int nt = 0; nt < 4; ++nt) {
    int n = n0 + nt*16 + l;
    float bini = (z==0) ? bq[n]*QSCALE : (z==2) ? bv[n] : 0.0f;
    for (int mt = 0; mt < 4; ++mt)
      for (int j = 0; j < 8; ++j) acc[mt][nt][j] = bini;
  }
  for (int k0 = 0; k0 < N_STATE; k0 += 32) {
    v16h bf[4], af[4];
    for (int nt = 0; nt < 4; ++nt)
      bf[nt] = load_b_frag(W + (size_t)(n0+nt*16)*N_STATE + k0, N_STATE, l, hi);
    for (int mt = 0; mt < 4; ++mt)
      af[mt] = load_a_frag(xh + (size_t)(m0+mt*16)*N_STATE + k0, N_STATE, l, hi);
    for (int mt = 0; mt < 4; ++mt)
      for (int nt = 0; nt < 4; ++nt)
        acc[mt][nt] = wmma_f16(af[mt], bf[nt], acc[mt][nt]);
  }
  if (z == 2) {
    // C-frag: lane holds fixed n, 8 consecutive m => contiguous along t in vT.
    for (int mt = 0; mt < 4; ++mt)
      for (int nt = 0; nt < 4; ++nt) {
        int n = n0 + nt*16 + l;
        int hh = n >> 6, d = n & 63;
        int m = m0 + mt*16 + hi*8;
        int bb = m >> 11, s = m & 2047;
        v8h p;
        for (int j = 0; j < 8; ++j) p[j] = (_Float16)acc[mt][nt][j];
        *(v8h*)(vT + (((size_t)bb*N_HEAD + hh)*HD + d)*SEQ + s) = p;
      }
  } else {
    _Float16* out = (z==0) ? qh : kh;
    for (int mt = 0; mt < 4; ++mt)
      for (int nt = 0; nt < 4; ++nt) {
        int n  = n0 + nt*16 + l;
        int mb = m0 + mt*16 + hi*8;
        for (int j = 0; j < 8; ++j)
          out[(size_t)(mb+j)*N_STATE + n] = (_Float16)acc[mt][nt][j];
      }
  }
}

// ---------------- causal flash attention, one wave per 16 queries ----------
// Computes S^T = K*Q^T (softmax rows lane-local), then O^T = V^T * P^T.
__global__ __launch_bounds__(128)
void flash_attn_kernel(const _Float16* __restrict__ qh,
                       const _Float16* __restrict__ kh,
                       const _Float16* __restrict__ vT,
                       _Float16* __restrict__ attn) {
  __shared__ _Float16 Plds[4][16][32];   // [wave][s_local][t_local]
  const int lane = threadIdx.x & 31, wave = threadIdx.x >> 5;
  const int l = lane & 15, hi = lane >> 4;
  const int bh = blockIdx.y, bb = bh >> 4, hh = bh & 15;
  const int s0 = (blockIdx.x*4 + wave)*16;
  const _Float16* qb = qh + ((size_t)bb*SEQ)*N_STATE + hh*HD;
  const _Float16* kb = kh + ((size_t)bb*SEQ)*N_STATE + hh*HD;
  const _Float16* vb = vT + (((size_t)bb*N_HEAD + hh)*HD)*SEQ;

  v16h qf[2];                                     // Q as B operand, d0 = 0,32
  for (int c = 0; c < 2; ++c)
    qf[c] = load_b_frag(qb + (size_t)s0*N_STATE + c*32, N_STATE, l, hi);

  float m_run = -INFINITY, l_run = 0.0f;
  v8f o[4];
  for (int dt = 0; dt < 4; ++dt)
    for (int j = 0; j < 8; ++j) o[dt][j] = 0.0f;

  const int s = s0 + l;
  const int nsteps = (s0 + 16 + 31) >> 5;         // 32-key steps, causal bound
  for (int step = 0; step < nsteps; ++step) {
    const int t0 = step*32;
    v8f st[2];
    for (int tt = 0; tt < 2; ++tt)
      for (int j = 0; j < 8; ++j) st[tt][j] = 0.0f;
    for (int tt = 0; tt < 2; ++tt)
      for (int c = 0; c < 2; ++c) {
        v16h kf = load_a_frag(kb + (size_t)(t0+tt*16)*N_STATE + c*32,
                              N_STATE, l, hi);
        st[tt] = wmma_f16(kf, qf[c], st[tt]);     // S^T tile: [t][s]
      }
    // causal mask + row max (row = fixed s = lane-local column of S^T)
    float mloc = -INFINITY;
    for (int tt = 0; tt < 2; ++tt)
      for (int r = 0; r < 8; ++r) {
        int t = t0 + tt*16 + hi*8 + r;
        float v = (t > s) ? -INFINITY : st[tt][r];
        st[tt][r] = v;
        mloc = fmaxf(mloc, v);
      }
    mloc = fmaxf(mloc, __shfl_xor(mloc, 16));
    const float m_new = fmaxf(m_run, mloc);
    const float alpha = __expf(m_run - m_new);
    float rs = 0.0f;
    for (int tt = 0; tt < 2; ++tt) {
      v8h p;
      for (int r = 0; r < 8; ++r) {
        float e = __expf(st[tt][r] - m_new);
        rs += e;
        p[r] = (_Float16)e;
      }
      *(v8h*)(&Plds[wave][l][tt*16 + hi*8]) = p;  // wave-private, in-order LDS
    }
    rs += __shfl_xor(rs, 16);
    l_run = l_run*alpha + rs;
    m_run = m_new;
    for (int dt = 0; dt < 4; ++dt)
      for (int j = 0; j < 8; ++j) o[dt][j] *= alpha;
    v16h pf = *(const v16h*)(&Plds[wave][l][hi*16]); // P^T as B operand
    for (int dt = 0; dt < 4; ++dt) {
      v16h vf = load_a_frag(vb + (size_t)(dt*16)*SEQ + t0, SEQ, l, hi);
      o[dt] = wmma_f16(vf, pf, o[dt]);            // O^T tile: [d][s]
    }
  }
  const float inv = 1.0f / l_run;
  for (int dt = 0; dt < 4; ++dt) {                // contiguous store into [s][D]
    v8h p;
    for (int j = 0; j < 8; ++j) p[j] = (_Float16)(o[dt][j]*inv);
    *(v8h*)(attn + ((size_t)bb*SEQ + s)*N_STATE + hh*HD + dt*16 + hi*8) = p;
  }
}

// ---------------- output projection, fp32 result ---------------------------
__global__ __launch_bounds__(128)
void out_proj_kernel(const _Float16* __restrict__ ah,
                     const _Float16* __restrict__ Woh,
                     const float* __restrict__ bo,
                     float* __restrict__ y) {
  const int lane = threadIdx.x & 31, wave = threadIdx.x >> 5;
  const int l = lane & 15, hi = lane >> 4;
  const int m0 = (blockIdx.x*4 + wave)*64;
  const int n0 = blockIdx.y*64;
  v8f acc[4][4];
  for (int nt = 0; nt < 4; ++nt) {
    float bini = bo[n0 + nt*16 + l];
    for (int mt = 0; mt < 4; ++mt)
      for (int j = 0; j < 8; ++j) acc[mt][nt][j] = bini;
  }
  for (int k0 = 0; k0 < N_STATE; k0 += 32) {
    v16h bf[4], af[4];
    for (int nt = 0; nt < 4; ++nt)
      bf[nt] = load_b_frag(Woh + (size_t)(n0+nt*16)*N_STATE + k0, N_STATE, l, hi);
    for (int mt = 0; mt < 4; ++mt)
      af[mt] = load_a_frag(ah + (size_t)(m0+mt*16)*N_STATE + k0, N_STATE, l, hi);
    for (int mt = 0; mt < 4; ++mt)
      for (int nt = 0; nt < 4; ++nt)
        acc[mt][nt] = wmma_f16(af[mt], bf[nt], acc[mt][nt]);
  }
  for (int mt = 0; mt < 4; ++mt)
    for (int nt = 0; nt < 4; ++nt) {
      int n  = n0 + nt*16 + l;
      int mb = m0 + mt*16 + hi*8;
      for (int j = 0; j < 8; ++j)
        y[(size_t)(mb+j)*N_STATE + n] = acc[mt][nt][j];
    }
}

extern "C" void kernel_launch(void* const* d_in, const int* in_sizes, int n_in,
                              void* d_out, int out_size, void* d_ws, size_t ws_size,
                              hipStream_t stream) {
  const float* x  = (const float*)d_in[0];
  // d_in[1] is the causal additive mask; implemented analytically (t > s -> -inf)
  const float* Wq = (const float*)d_in[2];
  const float* bq = (const float*)d_in[3];
  const float* Wk = (const float*)d_in[4];
  const float* Wv = (const float*)d_in[5];
  const float* bv = (const float*)d_in[6];
  const float* Wo = (const float*)d_in[7];
  const float* bo = (const float*)d_in[8];
  float* out = (float*)d_out;

  char* ws = (char*)d_ws;                       // 48 MB layout
  _Float16* xh  = (_Float16*)(ws + ((size_t)0 << 20));
  _Float16* Wqh = (_Float16*)(ws + ((size_t)8  << 20));
  _Float16* Wkh = (_Float16*)(ws + ((size_t)10 << 20));
  _Float16* Wvh = (_Float16*)(ws + ((size_t)12 << 20));
  _Float16* Woh = (_Float16*)(ws + ((size_t)14 << 20));
  _Float16* qh  = (_Float16*)(ws + ((size_t)16 << 20));
  _Float16* kh  = (_Float16*)(ws + ((size_t)24 << 20));
  _Float16* vT  = (_Float16*)(ws + ((size_t)32 << 20));
  _Float16* ah  = (_Float16*)(ws + ((size_t)40 << 20));

  const float QSCALE = 0.35355339059327379f;    // 64^(-1/4) folded into Wq, Wk
  const int NX = M_TOT*N_STATE;
  const int NW = N_STATE*N_STATE;
  cvt_f32_to_f16<<<(NX+255)/256, 256, 0, stream>>>(x,  xh,  NX, 1.0f);
  cvt_f32_to_f16<<<(NW+255)/256, 256, 0, stream>>>(Wq, Wqh, NW, QSCALE);
  cvt_f32_to_f16<<<(NW+255)/256, 256, 0, stream>>>(Wk, Wkh, NW, QSCALE);
  cvt_f32_to_f16<<<(NW+255)/256, 256, 0, stream>>>(Wv, Wvh, NW, 1.0f);
  cvt_f32_to_f16<<<(NW+255)/256, 256, 0, stream>>>(Wo, Woh, NW, 1.0f);

  qkv_proj_kernel <<<dim3(16,16,3), 128, 0, stream>>>(xh, Wqh, Wkh, Wvh,
                                                      bq, bv, qh, kh, vT);
  flash_attn_kernel<<<dim3(32,32),   128, 0, stream>>>(qh, kh, vT, ah);
  out_proj_kernel <<<dim3(16,16),    128, 0, stream>>>(ah, Woh, bo, out);
}
